// DiTBlock_24567212933338
// MI455X (gfx1250) — matrix operand order repaired and checked by reading, not compiled, not run on hardware
//
#include <hip/hip_runtime.h>
#include <math.h>

typedef __bf16 bf16;
typedef __bf16 v16bf __attribute__((ext_vector_type(16)));
typedef float  v8f   __attribute__((ext_vector_type(8)));

union FragA { v16bf v; uint4 q[2]; };
union Acc   { v8f v; uint4 q[2]; float f[8]; };
union Pack8 { uint4 q; bf16 h[8]; };
union Pack4 { uint2 q; bf16 h[4]; };

__device__ inline v8f wmma_bf16(v16bf a, v16bf b, v8f c) {
  // D = A(16x32 bf16) * B(32x16 bf16) + C(16x16 f32)
  return __builtin_amdgcn_wmma_f32_16x16x32_bf16(
      /*neg_a=*/false, a, /*neg_b=*/false, b,
      /*c_mod=*/(short)0, c, /*reuse_a=*/false, /*reuse_b=*/false);
}

// ---------------------------------------------------------------------------
// Weight transpose + cast: W (K x N, f32 row-major) -> WT (N x K, bf16)
// ---------------------------------------------------------------------------
__global__ __launch_bounds__(256) void transpose_cast(
    const float* __restrict__ W, bf16* __restrict__ WT, int K, int N) {
  __shared__ float tile[32][33];
  int n0 = blockIdx.x * 32, k0 = blockIdx.y * 32;
  int tx = threadIdx.x, ty = threadIdx.y;  // 32 x 8
#pragma unroll
  for (int i = 0; i < 4; i++)
    tile[ty + 8 * i][tx] = W[(size_t)(k0 + ty + 8 * i) * N + n0 + tx];
  __syncthreads();
#pragma unroll
  for (int i = 0; i < 4; i++)
    WT[(size_t)(n0 + ty + 8 * i) * K + k0 + tx] = (bf16)tile[tx][ty + 8 * i];
}

// ---------------------------------------------------------------------------
// ss = cond(B x 1024) @ w(1024 x 2048) + b   (tiny GEMM, plain f32 FMA)
// ---------------------------------------------------------------------------
__global__ __launch_bounds__(256) void condproj_kernel(
    const float* __restrict__ cond, const float* __restrict__ w,
    const float* __restrict__ bias, float* __restrict__ ss) {
  int col = blockIdx.x * 256 + threadIdx.x;  // < 2048
  int b = blockIdx.y;
  const float* c = cond + (size_t)b * 1024;
  float acc = bias[col];
#pragma unroll 4
  for (int k = 0; k < 1024; k++) acc = fmaf(c[k], w[(size_t)k * 2048 + col], acc);
  ss[(size_t)b * 2048 + col] = acc;
}

// ---------------------------------------------------------------------------
// adaLN: out = layernorm(x_row) * (1 + scale) + shift   -> bf16
// ---------------------------------------------------------------------------
__global__ __launch_bounds__(256) void adaln_kernel(
    const float* __restrict__ x, const float* __restrict__ ss,
    bf16* __restrict__ out) {
  __shared__ float red[256];
  __shared__ float red2[256];
  int row = blockIdx.x;       // 0..8191
  int b = row >> 10;
  int t = threadIdx.x;
  const float* xr = x + (size_t)row * 1024;
  float4 v4 = *(const float4*)&xr[t * 4];
  float v[4] = {v4.x, v4.y, v4.z, v4.w};
  float s = v[0] + v[1] + v[2] + v[3];
  float s2 = v[0] * v[0] + v[1] * v[1] + v[2] * v[2] + v[3] * v[3];
  red[t] = s; red2[t] = s2;
  __syncthreads();
  for (int off = 128; off > 0; off >>= 1) {
    if (t < off) { red[t] += red[t + off]; red2[t] += red2[t + off]; }
    __syncthreads();
  }
  float mu = red[0] * (1.0f / 1024.0f);
  float var = red2[0] * (1.0f / 1024.0f) - mu * mu;
  float rs = rsqrtf(var + 1e-6f);
  const float* ssb = ss + (size_t)b * 2048;
  Pack4 p;
#pragma unroll
  for (int e = 0; e < 4; e++) {
    int f = t * 4 + e;
    float o = (v[e] - mu) * rs * (1.0f + ssb[f]) + ssb[1024 + f];
    p.h[e] = (bf16)o;
  }
  *(uint2*)&out[(size_t)row * 1024 + t * 4] = p.q;
}

// ---------------------------------------------------------------------------
// Generic bf16 GEMM:  C(MxN) = A(MxK bf16) @ BT(NxK bf16)^T + bias [+resid][GELU]
// block tile 128x128, BK=64, double-buffered LDS, 256 threads = 8 waves (2x4),
// wave tile 64x32.  Global loads for tile i+1 issue before compute of tile i;
// prefetch (global_prefetch_b8) two tiles ahead.
// ---------------------------------------------------------------------------
#define BKK 64
#define LDK 72  // padded LDS row stride (elements) -> conflict-free b128 reads

template <bool GELU, bool RESID, bool BF16OUT>
__global__ __launch_bounds__(256) void gemm_kernel(
    const bf16* __restrict__ A, const bf16* __restrict__ BT,
    const float* __restrict__ bias, const float* __restrict__ resid,
    void* __restrict__ Cout, int M, int N, int K) {
  extern __shared__ char gsmem[];
  bf16* As = (bf16*)gsmem;          // 2 buffers of 128*LDK
  bf16* Bs = As + 2 * 128 * LDK;    // 2 buffers of 128*LDK
  int t = threadIdx.x;
  int wave = t >> 5, lane = t & 31;
  int l15 = lane & 15;
  int hi = lane >> 4;
  int chA = hi * 8;    // A-frag K-chunk base (interleaved 8s per ISA A layout)
  int chB = hi * 16;   // unused; B frags load with chA, same k order as A
  int wm = (wave >> 2) * 64;
  int wn = (wave & 3) * 32;
  int m0 = blockIdx.y * 128;
  int n0 = blockIdx.x * 128;

  Acc acc[4][2];
#pragma unroll
  for (int i = 0; i < 4; i++)
#pragma unroll
    for (int j = 0; j < 2; j++)
#pragma unroll
      for (int e = 0; e < 8; e++) acc[i][j].f[e] = 0.0f;

  int lrow = t >> 1;          // 0..127
  int lcb = (t & 1) * 32;     // 0 or 32 (covers one 64B cacheline)
  const bf16* Arow = A + (size_t)(m0 + lrow) * K + lcb;
  const bf16* Brow = BT + (size_t)(n0 + lrow) * K + lcb;

  uint4 ra[4], rb[4];
  auto gload = [&](int k0) {
#pragma unroll
    for (int j = 0; j < 4; j++) {
      ra[j] = *(const uint4*)&Arow[k0 + j * 8];
      rb[j] = *(const uint4*)&Brow[k0 + j * 8];
    }
  };
  auto sstore = [&](int buf) {
    bf16* as = As + buf * (128 * LDK) + lrow * LDK + lcb;
    bf16* bs = Bs + buf * (128 * LDK) + lrow * LDK + lcb;
#pragma unroll
    for (int j = 0; j < 4; j++) {
      *(uint4*)&as[j * 8] = ra[j];
      *(uint4*)&bs[j * 8] = rb[j];
    }
  };

  int nIter = K / BKK;
  gload(0);
  sstore(0);
  for (int it = 0; it < nIter; ++it) {
    int cur = it & 1;
    __syncthreads();
    if (it + 1 < nIter) gload((it + 1) * BKK);
    if (it + 2 < nIter) {
      __builtin_prefetch(&Arow[(it + 2) * BKK], 0, 0);
      __builtin_prefetch(&Brow[(it + 2) * BKK], 0, 0);
    }
    const bf16* as = As + cur * (128 * LDK);
    const bf16* bs = Bs + cur * (128 * LDK);
#pragma unroll
    for (int ks = 0; ks < 2; ks++) {
      FragA af[4], bfr[2];
#pragma unroll
      for (int i = 0; i < 4; i++) {
        int m = wm + i * 16 + l15;
        af[i].q[0] = *(const uint4*)&as[m * LDK + ks * 32 + chA];
        af[i].q[1] = *(const uint4*)&as[m * LDK + ks * 32 + 16 + chA];
      }
#pragma unroll
      for (int j = 0; j < 2; j++) {
        int n = wn + j * 16 + l15;
        bfr[j].q[0] = *(const uint4*)&bs[n * LDK + ks * 32 + chA];
        bfr[j].q[1] = *(const uint4*)&bs[n * LDK + ks * 32 + 16 + chA];
      }
#pragma unroll
      for (int i = 0; i < 4; i++)
#pragma unroll
        for (int j = 0; j < 2; j++)
          acc[i][j].v = wmma_bf16(af[i].v, bfr[j].v, acc[i][j].v);
    }
    if (it + 1 < nIter) sstore(1 - cur);
  }

#pragma unroll
  for (int i = 0; i < 4; i++) {
#pragma unroll
    for (int j = 0; j < 2; j++) {
      int col = n0 + wn + j * 16 + l15;
      float bv = bias[col];
      int mbase = m0 + wm + i * 16 + hi * 8;
#pragma unroll
      for (int r = 0; r < 8; r++) {
        size_t idx = (size_t)(mbase + r) * N + col;
        float v = acc[i][j].f[r] + bv;
        if (RESID) v += resid[idx];
        if (GELU) v = 0.5f * v * (1.0f + erff(v * 0.70710678118654752f));
        if (BF16OUT) ((bf16*)Cout)[idx] = (bf16)v;
        else         ((float*)Cout)[idx] = v;
      }
    }
  }
}

// ---------------------------------------------------------------------------
// Flash attention: one block per (b, h, 128-query-row block).
// qkv: [B,N,3072] bf16 (q|k|v each [h*64+d]).  ctx: [B,N,1024] bf16.
// Per key block of 128: S = q k^T (WMMA, S^T -> LDS f32), online softmax,
// O^T = V^T P^T (WMMA) with per-lane scalar alpha rescale.
// ---------------------------------------------------------------------------
#define QS 72    // q/k LDS row stride (elements)
#define VS 136   // vT LDS row stride
#define STS 132  // S^T LDS row stride (f32)
#define PS 136   // P LDS row stride

__global__ __launch_bounds__(256) void attn_kernel(
    const bf16* __restrict__ qkv, bf16* __restrict__ ctx) {
  extern __shared__ char smem[];
  bf16* qT = (bf16*)smem;                 // 128*72
  bf16* kT = qT + 128 * QS;               // 128*72
  bf16* vT = kT + 128 * QS;               // 64*136
  float* ST = (float*)(vT + 64 * VS);     // 128*132 f32
  bf16* Pm = (bf16*)(ST + 128 * STS);     // 128*136
  float* mrun = (float*)(Pm + 128 * PS);  // 128
  float* lrun = mrun + 128;               // 128
  float* abuf = lrun + 128;               // 128

  int t = threadIdx.x;
  int wave = t >> 5, lane = t & 31;
  int l15 = lane & 15, hi = lane >> 4;
  int chA = hi * 8, chB = hi * 16;
  int bh = blockIdx.y;
  int b = bh >> 4, h = bh & 15;
  int n0 = blockIdx.x * 128;
  const size_t RS = 3072;
  const float scale = 0.125f;  // 64^-0.5

  int lr = t >> 1;
  int lcb = (t & 1) * 32;

  // load q tile
  {
    const bf16* qb = qkv + ((size_t)b * 1024 + n0) * RS + h * 64;
#pragma unroll
    for (int j = 0; j < 4; j++) {
      int c = lcb + j * 8;
      *(uint4*)&qT[lr * QS + c] = *(const uint4*)&qb[(size_t)lr * RS + c];
    }
  }
  if (t < 128) { mrun[t] = -INFINITY; lrun[t] = 0.0f; }
  __syncthreads();

  FragA qf[2];
  {
    int m = wave * 16 + l15;
#pragma unroll
    for (int ks = 0; ks < 2; ks++) {
      qf[ks].q[0] = *(const uint4*)&qT[m * QS + ks * 32 + chA];
      qf[ks].q[1] = *(const uint4*)&qT[m * QS + ks * 32 + 16 + chA];
    }
  }

  Acc of[4];
#pragma unroll
  for (int i = 0; i < 4; i++)
#pragma unroll
    for (int e = 0; e < 8; e++) of[i].f[e] = 0.0f;

  for (int kb = 0; kb < 1024; kb += 128) {
    __syncthreads();
    {  // load k tile (row-major) and v tile (transposed) for this key block
      const bf16* kb_p = qkv + ((size_t)b * 1024 + kb) * RS + 1024 + h * 64;
      const bf16* vb_p = qkv + ((size_t)b * 1024 + kb) * RS + 2048 + h * 64;
      if (kb + 128 < 1024) {
        __builtin_prefetch(&kb_p[(size_t)(lr + 128) * RS + lcb], 0, 0);
        __builtin_prefetch(&vb_p[(size_t)(lr + 128) * RS + lcb], 0, 0);
      }
#pragma unroll
      for (int j = 0; j < 4; j++) {
        int c = lcb + j * 8;
        *(uint4*)&kT[lr * QS + c] = *(const uint4*)&kb_p[(size_t)lr * RS + c];
        Pack8 pv;
        pv.q = *(const uint4*)&vb_p[(size_t)lr * RS + c];
#pragma unroll
        for (int e = 0; e < 8; e++) vT[(c + e) * VS + lr] = pv.h[e];
      }
    }
    __syncthreads();
    // S = q @ k^T for this wave's 16 rows x 128 keys, store S^T to LDS
#pragma unroll
    for (int nk = 0; nk < 8; nk++) {
      int kr = nk * 16 + l15;
      FragA kf[2];
#pragma unroll
      for (int ks = 0; ks < 2; ks++) {
        kf[ks].q[0] = *(const uint4*)&kT[kr * QS + ks * 32 + chA];
        kf[ks].q[1] = *(const uint4*)&kT[kr * QS + ks * 32 + 16 + chA];
      }
      Acc s;
#pragma unroll
      for (int e = 0; e < 8; e++) s.f[e] = 0.0f;
      s.v = wmma_bf16(qf[0].v, kf[0].v, s.v);
      s.v = wmma_bf16(qf[1].v, kf[1].v, s.v);
      int key = nk * 16 + l15;
      int q0 = wave * 16 + hi * 8;
      *(uint4*)&ST[key * STS + q0] = s.q[0];
      *(uint4*)&ST[key * STS + q0 + 4] = s.q[1];
    }
    __syncthreads();
    // online softmax over this key block (2 threads per query row)
    {
      int r = t >> 1;
      int half = t & 1;
      int koff = half * 64;
      float mb = -INFINITY;
      for (int j = 0; j < 64; j++) mb = fmaxf(mb, ST[(koff + j) * STS + r]);
      mb = fmaxf(mb, __shfl_xor(mb, 1));
      mb *= scale;
      float mold = mrun[r];
      float mnew = fmaxf(mold, mb);
      float alpha = __expf(mold - mnew);
      float lsum = 0.0f;
      for (int j0 = 0; j0 < 64; j0 += 8) {
        Pack8 p8;
#pragma unroll
        for (int e = 0; e < 8; e++) {
          float sv = ST[(koff + j0 + e) * STS + r] * scale;
          float p = __expf(sv - mnew);
          lsum += p;
          p8.h[e] = (bf16)p;
        }
        *(uint4*)&Pm[r * PS + koff + j0] = p8.q;
      }
      lsum += __shfl_xor(lsum, 1);
      if (half == 0) {
        lrun[r] = lrun[r] * alpha + lsum;
        mrun[r] = mnew;
        abuf[r] = alpha;
      }
    }
    __syncthreads();
    // O^T += V^T @ P^T  (A = vT rows, B = P rows); rescale by alpha first
    float alpha_s = abuf[wave * 16 + l15];
#pragma unroll
    for (int i = 0; i < 4; i++)
#pragma unroll
      for (int e = 0; e < 8; e++) of[i].f[e] *= alpha_s;
    int qrow = wave * 16 + l15;
#pragma unroll
    for (int ks = 0; ks < 4; ks++) {
      FragA pf;
      pf.q[0] = *(const uint4*)&Pm[qrow * PS + ks * 32 + chA];
      pf.q[1] = *(const uint4*)&Pm[qrow * PS + ks * 32 + 16 + chA];
      FragA vf[4];
#pragma unroll
      for (int i = 0; i < 4; i++) {
        int d = i * 16 + l15;
        vf[i].q[0] = *(const uint4*)&vT[d * VS + ks * 32 + chA];
        vf[i].q[1] = *(const uint4*)&vT[d * VS + ks * 32 + 16 + chA];
      }
#pragma unroll
      for (int i = 0; i < 4; i++)
        of[i].v = wmma_bf16(vf[i].v, pf.v, of[i].v);
    }
  }
  // finalize: divide by l, write ctx (contiguous 8 x bf16 per fragment)
  float linv = 1.0f / lrun[wave * 16 + l15];
  int n = n0 + wave * 16 + l15;
  bf16* cb = ctx + ((size_t)b * 1024 + n) * 1024 + h * 64;
#pragma unroll
  for (int i = 0; i < 4; i++) {
    Pack8 p8;
#pragma unroll
    for (int e = 0; e < 8; e++) p8.h[e] = (bf16)(of[i].f[e] * linv);
    *(uint4*)&cb[i * 16 + hi * 8] = p8.q;
  }
}

// ---------------------------------------------------------------------------
extern "C" void kernel_launch(void* const* d_in, const int* in_sizes, int n_in,
                              void* d_out, int out_size, void* d_ws, size_t ws_size,
                              hipStream_t stream) {
  const float* x      = (const float*)d_in[0];
  const float* cond   = (const float*)d_in[1];
  const float* w_ada1 = (const float*)d_in[2];
  const float* b_ada1 = (const float*)d_in[3];
  const float* w_qkv  = (const float*)d_in[4];
  const float* b_qkv  = (const float*)d_in[5];
  const float* w_proj = (const float*)d_in[6];
  const float* b_proj = (const float*)d_in[7];
  const float* w_ada2 = (const float*)d_in[8];
  const float* b_ada2 = (const float*)d_in[9];
  const float* w_fc1  = (const float*)d_in[10];
  const float* b_fc1  = (const float*)d_in[11];
  const float* w_fc2  = (const float*)d_in[12];
  const float* b_fc2  = (const float*)d_in[13];
  float* out = (float*)d_out;

  char* ws = (char*)d_ws;
  size_t off = 0;
  auto alloc = [&](size_t bytes) -> char* {
    char* p = ws + off;
    off += (bytes + 255) & ~(size_t)255;
    return p;
  };
  float* ss1   = (float*)alloc((size_t)8 * 2048 * 4);
  float* ss2   = (float*)alloc((size_t)8 * 2048 * 4);
  bf16* h1     = (bf16*)alloc((size_t)8192 * 1024 * 2);
  bf16* wqkvT  = (bf16*)alloc((size_t)3072 * 1024 * 2);
  bf16* wprojT = (bf16*)alloc((size_t)1024 * 1024 * 2);
  bf16* wfc1T  = (bf16*)alloc((size_t)4096 * 1024 * 2);
  bf16* wfc2T  = (bf16*)alloc((size_t)1024 * 4096 * 2);
  bf16* qkvb   = (bf16*)alloc((size_t)8192 * 3072 * 2);
  bf16* ctx    = (bf16*)alloc((size_t)8192 * 1024 * 2);
  float* x1    = (float*)alloc((size_t)8192 * 1024 * 4);
  bf16* h2     = (bf16*)alloc((size_t)8192 * 1024 * 2);
  bf16* g      = (bf16*)alloc((size_t)8192 * 4096 * 2);

  dim3 blk256(256);
  dim3 tblk(32, 8);
  transpose_cast<<<dim3(3072 / 32, 1024 / 32), tblk, 0, stream>>>(w_qkv, wqkvT, 1024, 3072);
  transpose_cast<<<dim3(1024 / 32, 1024 / 32), tblk, 0, stream>>>(w_proj, wprojT, 1024, 1024);
  transpose_cast<<<dim3(4096 / 32, 1024 / 32), tblk, 0, stream>>>(w_fc1, wfc1T, 1024, 4096);
  transpose_cast<<<dim3(1024 / 32, 4096 / 32), tblk, 0, stream>>>(w_fc2, wfc2T, 4096, 1024);
  condproj_kernel<<<dim3(8, 8), blk256, 0, stream>>>(cond, w_ada1, b_ada1, ss1);
  condproj_kernel<<<dim3(8, 8), blk256, 0, stream>>>(cond, w_ada2, b_ada2, ss2);

  size_t gemm_lds = (size_t)4 * 128 * LDK * sizeof(bf16);  // 2 bufs x (A+B)

  adaln_kernel<<<8192, blk256, 0, stream>>>(x, ss1, h1);
  gemm_kernel<false, false, true><<<dim3(3072 / 128, 8192 / 128), blk256, gemm_lds, stream>>>(
      h1, wqkvT, b_qkv, nullptr, qkvb, 8192, 3072, 1024);

  size_t attn_lds = (size_t)(128 * QS + 128 * QS + 64 * VS) * 2 +
                    (size_t)128 * STS * 4 + (size_t)128 * PS * 2 + 3 * 128 * 4;
  attn_kernel<<<dim3(8, 128), blk256, attn_lds, stream>>>(qkvb, ctx);

  gemm_kernel<false, true, false><<<dim3(1024 / 128, 8192 / 128), blk256, gemm_lds, stream>>>(
      ctx, wprojT, b_proj, x, x1, 8192, 1024, 1024);
  adaln_kernel<<<8192, blk256, 0, stream>>>(x1, ss2, h2);
  gemm_kernel<true, false, true><<<dim3(4096 / 128, 8192 / 128), blk256, gemm_lds, stream>>>(
      h2, wfc1T, b_fc1, nullptr, g, 8192, 4096, 1024);
  gemm_kernel<false, true, false><<<dim3(1024 / 128, 8192 / 128), blk256, gemm_lds, stream>>>(
      g, wfc2T, b_fc2, x1, out, 8192, 1024, 4096);
}
